// DisBlock_13958643712815
// MI455X (gfx1250) — matrix operand order repaired and checked by reading, not compiled, not run on hardware
//
#include <hip/hip_runtime.h>
#include <hip/hip_bf16.h>
#include <stdint.h>

// CDNA5 / gfx1250: wave32, v_wmma_f32_16x16x32_bf16, async global->LDS (ASYNCcnt),
// double-buffered LDS with single barrier per K-step.
typedef __attribute__((ext_vector_type(16))) __bf16 v16bf;
typedef __attribute__((ext_vector_type(8)))  float  v8f;

#define EPSV 1e-5f

// ---------------------------------------------------------------------------
// zero-fill (u32 granularity) -- re-establishes padding halos each call
// ---------------------------------------------------------------------------
__global__ void fill_zero_u32(unsigned* __restrict__ p, long n) {
    long i = (long)blockIdx.x * blockDim.x + threadIdx.x;
    if (i < n) p[i] = 0u;
}

// ---------------------------------------------------------------------------
// f32 -> bf16 cast (1x1 weights: k = c ordering is already correct)
// ---------------------------------------------------------------------------
__global__ void cast_f32_to_bf16(const float* __restrict__ in,
                                 __bf16* __restrict__ out, long n) {
    long i = (long)blockIdx.x * blockDim.x + threadIdx.x;
    if (i < n) out[i] = (__bf16)in[i];
}

// ---------------------------------------------------------------------------
// 3x3 weight repack: OIHW f32 -> bf16 [N][K] with K ordered k = r*512 + c
// ---------------------------------------------------------------------------
__global__ void pack_w3(const float* __restrict__ w, __bf16* __restrict__ wb, long total) {
    long i = (long)blockIdx.x * blockDim.x + threadIdx.x;
    if (i >= total) return;
    int n = (int)(i / 4608);
    int t = (int)(i % 4608);
    int c = t / 9, r = t % 9;
    wb[(long)n * 4608 + r * 512 + c] = (__bf16)w[i];
}

// ---------------------------------------------------------------------------
// NCHW f32 x[32,512,32,32] -> padded NHWC bf16 xn[32,34,34,512] (tiled transpose)
// grid (16, 32, 32), block (32,8)
// ---------------------------------------------------------------------------
__global__ void transpose_pad_x(const float* __restrict__ X, __bf16* __restrict__ Xn) {
    __shared__ float tile[32][33];
    const int b = blockIdx.z, c0 = blockIdx.x * 32, s0 = blockIdx.y * 32;
    const int tx = threadIdx.x;
    const float* src = X + ((long)b * 512 + c0) * 1024 + s0;
    #pragma unroll
    for (int j = threadIdx.y; j < 32; j += 8)
        tile[j][tx] = src[(long)j * 1024 + tx];          // coalesced in s
    __syncthreads();
    #pragma unroll
    for (int j = threadIdx.y; j < 32; j += 8) {
        int s = s0 + j, h = s >> 5, w = s & 31;
        Xn[((long)(b * 34 + h + 1) * 34 + (w + 1)) * 512 + c0 + tx] =
            (__bf16)tile[tx][j];                          // coalesced in c
    }
}

// ---------------------------------------------------------------------------
// relu(bn(Y)) : NCHW f32 [32,512,16,16] -> padded NHWC bf16 [32,18,18,512]
// grid (16, 8, 32), block (32,8)
// ---------------------------------------------------------------------------
__global__ void bn_relu_tr(const float* __restrict__ Y, const float* __restrict__ ss,
                           __bf16* __restrict__ P) {
    __shared__ float tile[32][33];
    const int b = blockIdx.z, c0 = blockIdx.x * 32, s0 = blockIdx.y * 32;
    const int tx = threadIdx.x;
    #pragma unroll
    for (int j = threadIdx.y; j < 32; j += 8) {
        int c = c0 + j;
        float v = Y[((long)(b * 512 + c)) * 256 + s0 + tx];
        tile[j][tx] = fmaxf(v * ss[c] + ss[512 + c], 0.f);
    }
    __syncthreads();
    #pragma unroll
    for (int j = threadIdx.y; j < 32; j += 8) {
        int s = s0 + j, oh = s >> 4, ow = s & 15;
        P[((long)(b * 18 + oh + 1) * 18 + (ow + 1)) * 512 + c0 + tx] =
            (__bf16)tile[tx][j];
    }
}

// ---------------------------------------------------------------------------
// h = relu(bn(Ya)+bn(Yb)) -> padded NHWC bf16 [32,18,18,512] + NCHW f32 residual
// grid (16, 8, 32), block (32,8)
// ---------------------------------------------------------------------------
__global__ void fuse_tr(const float* __restrict__ Ya, const float* __restrict__ ssa,
                        const float* __restrict__ Yb, const float* __restrict__ ssb,
                        __bf16* __restrict__ Hn, float* __restrict__ Hf) {
    __shared__ float tile[32][33];
    const int b = blockIdx.z, c0 = blockIdx.x * 32, s0 = blockIdx.y * 32;
    const int tx = threadIdx.x;
    #pragma unroll
    for (int j = threadIdx.y; j < 32; j += 8) {
        int c = c0 + j;
        long idx = ((long)(b * 512 + c)) * 256 + s0 + tx;
        float v = fmaxf(Ya[idx] * ssa[c] + ssa[512 + c] +
                        Yb[idx] * ssb[c] + ssb[512 + c], 0.f);
        Hf[idx] = v;                                      // NCHW residual keep
        tile[j][tx] = v;
    }
    __syncthreads();
    #pragma unroll
    for (int j = threadIdx.y; j < 32; j += 8) {
        int s = s0 + j, oh = s >> 4, ow = s & 15;
        Hn[((long)(b * 18 + oh + 1) * 18 + (ow + 1)) * 512 + c0 + tx] =
            (__bf16)tile[tx][j];
    }
}

// ---------------------------------------------------------------------------
// spatial mean per (b,c)
// ---------------------------------------------------------------------------
__global__ void mean_hw(const float* __restrict__ X, float* __restrict__ out, int HW) {
    __shared__ float red[256];
    long base = (long)blockIdx.x * HW;
    float s = 0.f;
    for (int i = threadIdx.x; i < HW; i += blockDim.x) s += X[base + i];
    red[threadIdx.x] = s;
    __syncthreads();
    for (int off = 128; off > 0; off >>= 1) {
        if ((int)threadIdx.x < off) red[threadIdx.x] += red[threadIdx.x + off];
        __syncthreads();
    }
    if (threadIdx.x == 0) out[blockIdx.x] = red[0] / (float)HW;
}

// ---------------------------------------------------------------------------
// batch-norm stats over (B=32, HW=256) per channel; fused scale/shift out
// ---------------------------------------------------------------------------
__global__ void bn_stats(const float* __restrict__ Y, const float* __restrict__ gamma,
                         const float* __restrict__ beta, float* __restrict__ ss) {
    __shared__ float r1[256];
    __shared__ float r2[256];
    const int c = blockIdx.x;
    float s = 0.f, s2 = 0.f;
    for (int i = threadIdx.x; i < 32 * 256; i += 256) {
        int b = i >> 8, sp = i & 255;
        float v = Y[(((long)b << 9) + c) * 256 + sp];
        s += v; s2 += v * v;
    }
    r1[threadIdx.x] = s;
    r2[threadIdx.x] = s2;
    __syncthreads();
    for (int off = 128; off > 0; off >>= 1) {
        if ((int)threadIdx.x < off) {
            r1[threadIdx.x] += r1[threadIdx.x + off];
            r2[threadIdx.x] += r2[threadIdx.x + off];
        }
        __syncthreads();
    }
    if (threadIdx.x == 0) {
        const float inv_n = 1.f / 8192.f;
        float mean = r1[0] * inv_n;
        float var  = r2[0] * inv_n - mean * mean;
        float sc   = rsqrtf(var + EPSV) * gamma[c];
        ss[c]       = sc;
        ss[512 + c] = beta[c] - mean * sc;
    }
}

// ---------------------------------------------------------------------------
// dis[bc] = mean_s relu(bn(Y) + residual), HW = 256
// ---------------------------------------------------------------------------
__global__ void dis_fuse_mean(const float* __restrict__ Y, const float* __restrict__ ss,
                              const float* __restrict__ res, float* __restrict__ dis) {
    __shared__ float red[256];
    long bc = blockIdx.x;
    int c = (int)(bc & 511);
    long base = bc << 8;
    int t = threadIdx.x;
    red[t] = fmaxf(Y[base + t] * ss[c] + ss[512 + c] + res[base + t], 0.f);
    __syncthreads();
    for (int off = 128; off > 0; off >>= 1) {
        if (t < off) red[t] += red[t + off];
        __syncthreads();
    }
    if (t == 0) dis[bc] = red[0] * (1.f / 256.f);
}

// ---------------------------------------------------------------------------
// samples[s,b,c] = avg[b,c] + dis[b,c]*noise[s,b,c]   (BC = 16384 = 2^14)
// ---------------------------------------------------------------------------
__global__ void sample_kernel(const float* __restrict__ avg, const float* __restrict__ dis,
                              const float* __restrict__ noise, float* __restrict__ out,
                              long total) {
    long i = (long)blockIdx.x * blockDim.x + threadIdx.x;
    if (i >= total) return;
    long bc = i & 16383;
    out[i] = fmaf(dis[bc], noise[i], avg[bc]);
}

// ---------------------------------------------------------------------------
// Implicit-GEMM conv, fully async, double-buffered LDS, 1 barrier per K-step.
//   input: padded NHWC bf16 [B, HiP, WiP, Ci] (halo pre-zeroed)
//   weights: bf16 [N][K], K ordered k = r*Ci + c
//   M = B*OH*OW, N = Cout.  Block tile 128x128, K-step 32 (32 channels at one
//   fixed kernel position r; Ci % 32 == 0).  8 waves: 2x4 WMMA accs each.
// Step s: [s_wait_asynccnt 0; barrier; issue async(s+1) -> buf[(s+1)&1];
//          compute buf[s&1]].  Barrier completion implies (a) all waves' buf(s)
//          async-writes landed (each wave drains ASYNCcnt pre-signal) and
//          (b) all waves' ds-reads of buf(s+1) from step s-1 are done (waves
//          drain DScnt pre-signal) -- so the overwrite is WAR-safe and the
//          next step's loads fully overlap this step's 8 WMMAs.
// ---------------------------------------------------------------------------
template <int HiP, int WiP, int OH, int OW, int KH, int KW,
          int STRIDE, int POFF, int N, int K, int Ci>
__global__ __launch_bounds__(256) void conv_gemm_bf16(
    const __bf16* __restrict__ Ain, const __bf16* __restrict__ Wk,
    float* __restrict__ Y)
{
    constexpr int OHW    = OH * OW;
    constexpr int CSTEPS = Ci / 32;            // 16
    constexpr int NSTEPS = KH * KW * CSTEPS;   // 144 (3x3) or 16 (1x1)
    constexpr int BUFE   = 128 * 32;           // elements per LDS buffer

    __shared__ __align__(16) __bf16 As[2 * BUFE];   // [buf][m][k_local]
    __shared__ __align__(16) __bf16 Bs[2 * BUFE];   // [buf][n][k_local]

    const int tid  = threadIdx.x;
    const int lane = tid & 31;
    const int wave = tid >> 5;
    const int wm   = wave & 3;
    const int wn   = wave >> 2;
    const int lrow = lane & 15;
    const int lhi  = lane >> 4;

    const int tileM = blockIdx.x * 128;
    const int tileN = blockIdx.y * 128;

    // A-side: thread covers row a_mloc, 16 channels starting at a_k0
    const int a_mloc = tid >> 1;
    const int a_k0   = (tid & 1) * 16;
    const int m      = tileM + a_mloc;
    const int b_img  = m / OHW;
    const int sp     = m % OHW;
    const int ohs    = (sp / OW) * STRIDE + POFF;
    const int ows    = (sp % OW) * STRIDE + POFF;
    const __bf16* gAr0 = Ain + ((long)(b_img * HiP + ohs) * WiP + ows) * Ci + a_k0;
    const unsigned ldsA0 = (unsigned)(uintptr_t)(void*)&As[a_mloc * 32 + a_k0];
    const unsigned ldsA1 = ldsA0 + 16;

    // B-side: thread covers n-row bn_loc, quarter bq (8 channels)
    const int  bn_loc = tid >> 2;
    const int  bq     = (tid & 3) * 8;
    const unsigned ldsB0 = (unsigned)(uintptr_t)(void*)&Bs[bn_loc * 32 + bq];
    const unsigned ldsB1 = ldsB0 + 64 * 32 * 2;
    const __bf16* gB0 = Wk + (long)(tileN + bn_loc) * K + bq;
    const __bf16* gB1 = Wk + (long)(tileN + 64 + bn_loc) * K + bq;

    v8f acc[2][4] = {};

    auto issueStep = [&](int s) {
        const int r  = s / CSTEPS;                 // compile-time divisor
        const int cb = (s % CSTEPS) * 32;
        const int rh = r / KW, rw = r % KW;        // KW constant
        const unsigned boff = (unsigned)((s & 1) * (BUFE * 2));
        const __bf16* gAs  = gAr0 + (rh * WiP + rw) * Ci + cb;
        const __bf16* gBs0 = gB0 + s * 32;         // k linear in s
        const __bf16* gBs1 = gB1 + s * 32;
        asm volatile("global_load_async_to_lds_b128 %0, %1, off"
                     :: "v"(ldsA0 + boff),
                        "v"((unsigned long long)(uintptr_t)gAs) : "memory");
        asm volatile("global_load_async_to_lds_b128 %0, %1, off"
                     :: "v"(ldsA1 + boff),
                        "v"((unsigned long long)(uintptr_t)(gAs + 8)) : "memory");
        asm volatile("global_load_async_to_lds_b128 %0, %1, off"
                     :: "v"(ldsB0 + boff),
                        "v"((unsigned long long)(uintptr_t)gBs0) : "memory");
        asm volatile("global_load_async_to_lds_b128 %0, %1, off"
                     :: "v"(ldsB1 + boff),
                        "v"((unsigned long long)(uintptr_t)gBs1) : "memory");
    };

    issueStep(0);

    #pragma unroll 1
    for (int s = 0; s < NSTEPS; ++s) {
        asm volatile("s_wait_asynccnt 0x0" ::: "memory");
        __syncthreads();
        if (s + 1 < NSTEPS) issueStep(s + 1);      // overlaps compute below

        const int bo = (s & 1) * BUFE;
        union FB { uint4 q[2]; v16bf v; };
        FB fa[2], fb[4];
        #pragma unroll
        for (int mt = 0; mt < 2; ++mt) {
            int row = wm * 32 + mt * 16 + lrow;
            const __bf16* pp = &As[bo + row * 32 + lhi * 8];
            fa[mt].q[0] = *(const uint4*)pp;
            fa[mt].q[1] = *(const uint4*)(pp + 16);
        }
        #pragma unroll
        for (int nt = 0; nt < 4; ++nt) {
            int col = wn * 64 + nt * 16 + lrow;
            const __bf16* pp = &Bs[bo + col * 32 + lhi * 8];
            fb[nt].q[0] = *(const uint4*)pp;
            fb[nt].q[1] = *(const uint4*)(pp + 16);
        }
        #pragma unroll
        for (int mt = 0; mt < 2; ++mt)
            #pragma unroll
            for (int nt = 0; nt < 4; ++nt)
                acc[mt][nt] = __builtin_amdgcn_wmma_f32_16x16x32_bf16(
                    false, fa[mt].v, false, fb[nt].v,
                    (short)0, acc[mt][nt], false, false);
    }

    // epilogue: scatter accumulators to NCHW f32 output
    #pragma unroll
    for (int mt = 0; mt < 2; ++mt) {
        #pragma unroll
        for (int r = 0; r < 8; ++r) {
            int mm  = tileM + wm * 32 + mt * 16 + lhi * 8 + r;
            int bi  = mm / OHW;
            int spo = mm % OHW;
            #pragma unroll
            for (int nt = 0; nt < 4; ++nt) {
                int nn = tileN + wn * 64 + nt * 16 + lrow;
                Y[((long)bi * N + nn) * OHW + spo] = acc[mt][nt][r];
            }
        }
    }
}

// ---------------------------------------------------------------------------
// host driver
// ---------------------------------------------------------------------------
extern "C" void kernel_launch(void* const* d_in, const int* in_sizes, int n_in,
                              void* d_out, int out_size, void* d_ws, size_t ws_size,
                              hipStream_t stream) {
    (void)in_sizes; (void)n_in; (void)out_size; (void)ws_size;

    const float* x     = (const float*)d_in[0];
    const float* noise = (const float*)d_in[1];
    const float* w1a = (const float*)d_in[2];
    const float* g1a = (const float*)d_in[3];
    const float* b1a = (const float*)d_in[4];
    const float* w1b = (const float*)d_in[5];
    const float* g1b = (const float*)d_in[6];
    const float* b1b = (const float*)d_in[7];
    const float* ws1 = (const float*)d_in[8];
    const float* gs1 = (const float*)d_in[9];
    const float* bs1 = (const float*)d_in[10];
    const float* w2a = (const float*)d_in[11];
    const float* g2a = (const float*)d_in[12];
    const float* b2a = (const float*)d_in[13];
    const float* w2b = (const float*)d_in[14];
    const float* g2b = (const float*)d_in[15];
    const float* b2b = (const float*)d_in[16];
    float* out = (float*)d_out;

    const int  C   = 512;
    const long NY  = 4194304;                    // 32*512*16*16
    const long BC  = 16384;
    const long NS  = 100L * BC;
    const long W3  = (long)C * C * 9;            // 2,359,296
    const long W1  = (long)C * C;
    const long NXN = 32L * 34 * 34 * 512;        // 18,939,904 (padded x)
    const long NPN = 32L * 18 * 18 * 512;        //  5,308,416 (padded 16x16)

    char* p = (char*)d_ws;
    auto alloc = [&](size_t bytes) {
        char* r = p;
        p += (bytes + 255) & ~(size_t)255;
        return r;
    };
    __bf16* xn   = (__bf16*)alloc((size_t)NXN * 2);  // padded NHWC x
    __bf16* P1   = (__bf16*)alloc((size_t)NPN * 2);  // padded NHWC act (reused)
    __bf16* H1n  = (__bf16*)alloc((size_t)NPN * 2);  // padded NHWC resblock1 out
    __bf16* wb1a = (__bf16*)alloc((size_t)W3 * 2);
    __bf16* wb1b = (__bf16*)alloc((size_t)W3 * 2);
    __bf16* wb2a = (__bf16*)alloc((size_t)W3 * 2);
    __bf16* wb2b = (__bf16*)alloc((size_t)W3 * 2);
    __bf16* wbsc = (__bf16*)alloc((size_t)W1 * 2);
    float*  YA   = (float*)alloc((size_t)NY * 4);
    float*  YB   = (float*)alloc((size_t)NY * 4);
    float*  H1f  = (float*)alloc((size_t)NY * 4);
    float* ss1 = (float*)alloc(2 * C * 4);
    float* ss2 = (float*)alloc(2 * C * 4);
    float* ss3 = (float*)alloc(2 * C * 4);
    float* ss4 = (float*)alloc(2 * C * 4);
    float* ss5 = (float*)alloc(2 * C * 4);

    const dim3 gconv(64, 4);
    const dim3 btr(32, 8);

    // zero padded buffers (halos must be 0 every call)
    fill_zero_u32<<<(unsigned)(NXN / 2 / 256), 256, 0, stream>>>((unsigned*)xn, NXN / 2);
    fill_zero_u32<<<(unsigned)(NPN / 2 / 256), 256, 0, stream>>>((unsigned*)P1, NPN / 2);
    fill_zero_u32<<<(unsigned)(NPN / 2 / 256), 256, 0, stream>>>((unsigned*)H1n, NPN / 2);

    // repacks
    transpose_pad_x<<<dim3(16, 32, 32), btr, 0, stream>>>(x, xn);
    pack_w3<<<(unsigned)(W3 / 256), 256, 0, stream>>>(w1a, wb1a, W3);
    pack_w3<<<(unsigned)(W3 / 256), 256, 0, stream>>>(w1b, wb1b, W3);
    pack_w3<<<(unsigned)(W3 / 256), 256, 0, stream>>>(w2a, wb2a, W3);
    pack_w3<<<(unsigned)(W3 / 256), 256, 0, stream>>>(w2b, wb2b, W3);
    cast_f32_to_bf16<<<(unsigned)(W1 / 256), 256, 0, stream>>>(ws1, wbsc, W1);

    // avg -> out[0:16384]
    mean_hw<<<(unsigned)BC, 256, 0, stream>>>(x, out, 1024);

    // ---- resblock 1 (stride 2, 1x1 shortcut) ----
    conv_gemm_bf16<34,34,16,16,3,3,2,0,512,4608,512>
        <<<gconv, 256, 0, stream>>>(xn, wb1a, YA);
    bn_stats<<<C, 256, 0, stream>>>(YA, g1a, b1a, ss1);
    bn_relu_tr<<<dim3(16, 8, 32), btr, 0, stream>>>(YA, ss1, P1);

    conv_gemm_bf16<18,18,16,16,3,3,1,0,512,4608,512>
        <<<gconv, 256, 0, stream>>>(P1, wb1b, YB);
    bn_stats<<<C, 256, 0, stream>>>(YB, g1b, b1b, ss2);

    conv_gemm_bf16<34,34,16,16,1,1,2,1,512,512,512>
        <<<gconv, 256, 0, stream>>>(xn, wbsc, YA);
    bn_stats<<<C, 256, 0, stream>>>(YA, gs1, bs1, ss3);

    fuse_tr<<<dim3(16, 8, 32), btr, 0, stream>>>(YB, ss2, YA, ss3, H1n, H1f);

    // ---- resblock 2 (stride 1, identity shortcut) ----
    conv_gemm_bf16<18,18,16,16,3,3,1,0,512,4608,512>
        <<<gconv, 256, 0, stream>>>(H1n, wb2a, YA);
    bn_stats<<<C, 256, 0, stream>>>(YA, g2a, b2a, ss4);
    bn_relu_tr<<<dim3(16, 8, 32), btr, 0, stream>>>(YA, ss4, P1);

    conv_gemm_bf16<18,18,16,16,3,3,1,0,512,4608,512>
        <<<gconv, 256, 0, stream>>>(P1, wb2b, YB);
    bn_stats<<<C, 256, 0, stream>>>(YB, g2b, b2b, ss5);

    // dis -> out[16384:32768]
    dis_fuse_mean<<<(unsigned)BC, 256, 0, stream>>>(YB, ss5, H1f, out + BC);

    // samples -> out[32768:]
    sample_kernel<<<(unsigned)(NS / 256), 256, 0, stream>>>(out, out + BC, noise,
                                                            out + 2 * BC, NS);
}
